// GATModule_13005160972561
// MI455X (gfx1250) — matrix operand order repaired and checked by reading, not compiled, not run on hardware
//
#include <hip/hip_runtime.h>
#include <hip/hip_bf16.h>
#include <math.h>

typedef _Float16 half_t;
typedef __attribute__((ext_vector_type(16))) _Float16 v16h;
typedef __attribute__((ext_vector_type(8)))  float    v8f;
typedef __attribute__((ext_vector_type(4)))  unsigned uint4v;

#define ACT_NONE  0
#define ACT_LRELU 1
#define ACT_TANH  2

static const int Hdim = 512;
static const int HEAD = 8;
static const int Ddim = 64;
static const int Bsz  = 4096;
static const int NNB  = 8192;
static const int Sdim = 16;
static const int Kn   = 16;
static const int Gsz  = Bsz + NNB;   // 12288

// ---------------------------------------------------------------------------
// Weight conversion: f32 -> f16, pre-transposed to [N][K] layout for WMMA B.
// ---------------------------------------------------------------------------
__global__ void convert_weights(const float* __restrict__ projW,
                                const float* __restrict__ wsW,
                                const float* __restrict__ gatW,
                                const float* __restrict__ ffW,
                                half_t* projWt, half_t* wsWt,
                                half_t* gatWt, half_t* ffWt) {
    int i = blockIdx.x * blockDim.x + threadIdx.x;
    if (i < 512 * 512) {
        int j = i >> 9, k = i & 511;
        projWt[i] = (half_t)projW[k * 512 + j];
        return;
    }
    i -= 512 * 512;
    if (i < 2 * 512 * 512) {
        int h = i / (512 * 512); int r = i % (512 * 512);
        int j = r >> 9, k = r & 511;
        int e = j >> 6, d = j & 63;
        wsWt[h * 512 * 512 + r] =
            (half_t)wsW[(((size_t)h * 8 + e) * 512 + k) * 64 + d];
        return;
    }
    i -= 2 * 512 * 512;
    if (i < 2 * 8 * 64 * 128) {
        int h = i / (8 * 64 * 128); int r = i % (8 * 64 * 128);
        int e = r / (64 * 128);     int r2 = r % (64 * 128);
        int d = r2 >> 7;            int c = r2 & 127;
        gatWt[i] = (half_t)gatW[(((size_t)h * 8 + e) * 128 + c) * 64 + d];
        return;
    }
    i -= 2 * 8 * 64 * 128;
    if (i < 512 * 1024) {
        int j = i >> 10, k = i & 1023;
        ffWt[i] = (half_t)ffW[k * 512 + j];
    }
}

// ---------------------------------------------------------------------------
// Masked mean over S=16 (linearity: mean before projection). One block/node.
// ---------------------------------------------------------------------------
__global__ void masked_mean_kernel(const float* __restrict__ span_out,
                                   const int* __restrict__ span_mask,
                                   const float* __restrict__ nb_out,
                                   const int* __restrict__ nb_mask,
                                   half_t* __restrict__ xbar) {
    int n = blockIdx.x;
    const float* x; const int* m;
    if (n < Bsz) { x = span_out + (size_t)n * Sdim * Hdim; m = span_mask + (size_t)n * Sdim; }
    else { int q = n - Bsz; x = nb_out + (size_t)q * Sdim * Hdim; m = nb_mask + (size_t)q * Sdim; }
    int t = threadIdx.x;                 // 256 threads, 2 cols each
    float a0 = 0.f, a1 = 0.f, cnt = 0.f;
    for (int s = 0; s < Sdim; ++s) {
        int keep = (m[s] != 1);
        cnt += (float)keep;
        if (keep) {
            a0 += x[s * Hdim + t];
            a1 += x[s * Hdim + t + 256];
        }
    }
    float inv = 1.f / cnt;               // cnt >= 1 (mask[:,0]==0)
    xbar[(size_t)n * Hdim + t]       = (half_t)(a0 * inv);
    xbar[(size_t)n * Hdim + t + 256] = (half_t)(a1 * inv);
}

// ---------------------------------------------------------------------------
// f16 WMMA GEMM: C = act(A * Bt^T + bias). One wave = 32x64 output tile
// (2 A fragments x 4 B fragments = 8 WMMAs/k-step), software-pipelined:
// fragments for k+32 are loaded before the WMMAs for k are issued.
// A frag: lane row M=lane&15, K (lane>>4)*16 + 0..15
// B frag: lane col N=lane&15, K (lane>>4)*16 + 0..15
// ---------------------------------------------------------------------------
struct Frag { uint4v q0, q1; };

__device__ __forceinline__ v16h frag_v(const Frag& f) {
    union { Frag f; v16h v; } u; u.f = f; return u.v;
}
__device__ __forceinline__ Frag loadA(const half_t* p, int k, int hiA) {
    Frag f;
    f.q0 = *(const uint4v*)(p + k + hiA);        // K pairs 0,2,4,6
    f.q1 = *(const uint4v*)(p + k + hiA + 8);    // K pairs 8,10,12,14
    return f;
}
__device__ __forceinline__ Frag loadB(const half_t* p, int k, int hiB) {
    Frag f;
    f.q0 = *(const uint4v*)(p + k + hiB);
    f.q1 = *(const uint4v*)(p + k + hiB + 8);
    return f;
}

__global__ __launch_bounds__(128)
void gemm_wmma(const half_t* __restrict__ A, int lda, int aHeadStride,
               const half_t* __restrict__ Bt, int bHeadStride,
               const float* __restrict__ bias, int biasHeadStride,
               half_t* outH, float* outF, int ldc, int cHeadStride,
               const int* __restrict__ zeroRows,
               int M, int N, int K, int act) {
    int lane = threadIdx.x & 31;
    int wave = threadIdx.x >> 5;
    int e    = blockIdx.z;
    int mt   = blockIdx.y * 4 + wave;    // wave handles rows [mt*32, mt*32+32)
    if (mt * 32 >= M) return;
    int row0 = mt * 32 + (lane & 15);
    int hiA  = (lane >> 4) * 16;
    int hiB  = (lane >> 4) * 16;

    const half_t* Ap0 = A + (size_t)e * aHeadStride + (size_t)row0 * lda;
    const half_t* Ap1 = Ap0 + (size_t)16 * lda;
    int nbase = blockIdx.x * 64;
    const half_t* Bp0 = Bt + (size_t)e * bHeadStride + (size_t)(nbase + 0 * 16 + (lane & 15)) * K;
    const half_t* Bp1 = Bt + (size_t)e * bHeadStride + (size_t)(nbase + 1 * 16 + (lane & 15)) * K;
    const half_t* Bp2 = Bt + (size_t)e * bHeadStride + (size_t)(nbase + 2 * 16 + (lane & 15)) * K;
    const half_t* Bp3 = Bt + (size_t)e * bHeadStride + (size_t)(nbase + 3 * 16 + (lane & 15)) * K;

    v8f acc[2][4] = {};

    // pipeline prologue: fragments for k0 = 0
    Frag a0 = loadA(Ap0, 0, hiA), a1 = loadA(Ap1, 0, hiA);
    Frag b0 = loadB(Bp0, 0, hiB), b1 = loadB(Bp1, 0, hiB);
    Frag b2 = loadB(Bp2, 0, hiB), b3 = loadB(Bp3, 0, hiB);

    for (int k0 = 0; k0 + 32 < K; k0 += 32) {
        int kn = k0 + 32;
        // issue next iteration's loads before consuming current fragments
        Frag na0 = loadA(Ap0, kn, hiA), na1 = loadA(Ap1, kn, hiA);
        Frag nb0 = loadB(Bp0, kn, hiB), nb1 = loadB(Bp1, kn, hiB);
        Frag nb2 = loadB(Bp2, kn, hiB), nb3 = loadB(Bp3, kn, hiB);

        acc[0][0] = __builtin_amdgcn_wmma_f32_16x16x32_f16(false, frag_v(a0), false, frag_v(b0), (short)0, acc[0][0], false, false);
        acc[0][1] = __builtin_amdgcn_wmma_f32_16x16x32_f16(false, frag_v(a0), false, frag_v(b1), (short)0, acc[0][1], false, false);
        acc[0][2] = __builtin_amdgcn_wmma_f32_16x16x32_f16(false, frag_v(a0), false, frag_v(b2), (short)0, acc[0][2], false, false);
        acc[0][3] = __builtin_amdgcn_wmma_f32_16x16x32_f16(false, frag_v(a0), false, frag_v(b3), (short)0, acc[0][3], false, false);
        acc[1][0] = __builtin_amdgcn_wmma_f32_16x16x32_f16(false, frag_v(a1), false, frag_v(b0), (short)0, acc[1][0], false, false);
        acc[1][1] = __builtin_amdgcn_wmma_f32_16x16x32_f16(false, frag_v(a1), false, frag_v(b1), (short)0, acc[1][1], false, false);
        acc[1][2] = __builtin_amdgcn_wmma_f32_16x16x32_f16(false, frag_v(a1), false, frag_v(b2), (short)0, acc[1][2], false, false);
        acc[1][3] = __builtin_amdgcn_wmma_f32_16x16x32_f16(false, frag_v(a1), false, frag_v(b3), (short)0, acc[1][3], false, false);

        a0 = na0; a1 = na1; b0 = nb0; b1 = nb1; b2 = nb2; b3 = nb3;
    }
    // pipeline epilogue: last k-step
    acc[0][0] = __builtin_amdgcn_wmma_f32_16x16x32_f16(false, frag_v(a0), false, frag_v(b0), (short)0, acc[0][0], false, false);
    acc[0][1] = __builtin_amdgcn_wmma_f32_16x16x32_f16(false, frag_v(a0), false, frag_v(b1), (short)0, acc[0][1], false, false);
    acc[0][2] = __builtin_amdgcn_wmma_f32_16x16x32_f16(false, frag_v(a0), false, frag_v(b2), (short)0, acc[0][2], false, false);
    acc[0][3] = __builtin_amdgcn_wmma_f32_16x16x32_f16(false, frag_v(a0), false, frag_v(b3), (short)0, acc[0][3], false, false);
    acc[1][0] = __builtin_amdgcn_wmma_f32_16x16x32_f16(false, frag_v(a1), false, frag_v(b0), (short)0, acc[1][0], false, false);
    acc[1][1] = __builtin_amdgcn_wmma_f32_16x16x32_f16(false, frag_v(a1), false, frag_v(b1), (short)0, acc[1][1], false, false);
    acc[1][2] = __builtin_amdgcn_wmma_f32_16x16x32_f16(false, frag_v(a1), false, frag_v(b2), (short)0, acc[1][2], false, false);
    acc[1][3] = __builtin_amdgcn_wmma_f32_16x16x32_f16(false, frag_v(a1), false, frag_v(b3), (short)0, acc[1][3], false, false);

    // epilogue
    int rl = (lane >> 4) * 8;
#pragma unroll
    for (int mi = 0; mi < 2; ++mi) {
#pragma unroll
        for (int t = 0; t < 4; ++t) {
            int col  = nbase + t * 16 + (lane & 15);
            float bv = bias ? bias[e * biasHeadStride + col] : 0.f;
#pragma unroll
            for (int r = 0; r < 8; ++r) {
                float v = acc[mi][t][r] + bv;
                if (act == ACT_LRELU)      v = v > 0.f ? v : 0.01f * v;
                else if (act == ACT_TANH)  v = tanhf(v);
                int orow = mt * 32 + mi * 16 + rl + r;
                if (zeroRows && zeroRows[orow]) v = 0.f;
                size_t ci = (size_t)orow * ldc + (size_t)e * cHeadStride + col;
                if (outH) outH[ci] = (half_t)v;
                if (outF) outF[ci] = v;
            }
        }
    }
}

// ---------------------------------------------------------------------------
// GAT attention: one wave per (node, head). Lane owns d = 2*lane, 2*lane+1.
// comb[n][e*128 + 0..63] = mixed, comb[n][e*128 + 64..127] = node.
// ---------------------------------------------------------------------------
__global__ __launch_bounds__(256)
void attn_kernel(const half_t* __restrict__ ch,
                 const long long* __restrict__ gmap,
                 half_t* __restrict__ comb,
                 int* __restrict__ emptyFlags, int n_sz) {
    int n    = blockIdx.x;
    int e    = threadIdx.x >> 5;      // 8 waves = 8 heads
    int lane = threadIdx.x & 31;

    int idxlo = 0, padi = 0;
    if (lane < Kn) {
        long long gi = gmap[(size_t)n * Kn + lane];
        padi  = (gi < 0) ? 1 : 0;
        idxlo = padi ? 0 : (int)gi;
    }
    int pc = padi;
#pragma unroll
    for (int off = 16; off >= 1; off >>= 1) pc += __shfl_xor(pc, off, 32);
    bool empty = (pc == Kn);
    if (threadIdx.x == 0) emptyFlags[n] = empty ? 1 : 0;

    const half_t* np = ch + (size_t)n * Hdim + e * Ddim + lane * 2;
    float n0 = (float)np[0], n1 = (float)np[1];

    float c0[16], c1[16], s[16];
#pragma unroll
    for (int k = 0; k < Kn; ++k) {
        int safe = __shfl(idxlo, k, 32);
        int pk   = __shfl(padi, k, 32);
        const half_t* cp = ch + (size_t)safe * Hdim + e * Ddim + lane * 2;
        float a0 = (float)cp[0], a1 = (float)cp[1];
        c0[k] = a0; c1[k] = a1;
        float p = n0 * a0 + n1 * a1;
#pragma unroll
        for (int off = 16; off >= 1; off >>= 1) p += __shfl_xor(p, off, 32);
        s[k] = (!empty && pk) ? -1.0e9f : p;
    }
    float mx = s[0];
#pragma unroll
    for (int k = 1; k < Kn; ++k) mx = fmaxf(mx, s[k]);
    float sum = 0.f;
#pragma unroll
    for (int k = 0; k < Kn; ++k) { s[k] = expf(s[k] - mx); sum += s[k]; }
    float inv = 1.f / sum;
    float m0 = 0.f, m1 = 0.f;
#pragma unroll
    for (int k = 0; k < Kn; ++k) { float a = s[k] * inv; m0 += a * c0[k]; m1 += a * c1[k]; }

    half_t* cb = comb + (size_t)n * (HEAD * 128) + e * 128;
    cb[lane * 2]          = (half_t)m0;
    cb[lane * 2 + 1]      = (half_t)m1;
    cb[64 + lane * 2]     = (half_t)n0;
    cb[64 + lane * 2 + 1] = (half_t)n1;
}

// ---------------------------------------------------------------------------
// Build ff input: [span_hidden | where(empty, b_hidden, c)] as f16.
// ---------------------------------------------------------------------------
__global__ void concat_kernel(const float* __restrict__ span_hidden,
                              const half_t* __restrict__ all_h_h,
                              const half_t* __restrict__ c2_h,
                              const int* __restrict__ emptyFlags,
                              half_t* __restrict__ ffin) {
    int i = blockIdx.x * blockDim.x + threadIdx.x;
    int n = i >> 10, col = i & 1023;
    half_t v;
    if (col < 512) v = (half_t)span_hidden[(size_t)n * 512 + col];
    else {
        int j = col - 512;
        v = emptyFlags[n] ? all_h_h[(size_t)n * 512 + j] : c2_h[(size_t)n * 512 + j];
    }
    ffin[i] = v;
}

// ---------------------------------------------------------------------------
extern "C" void kernel_launch(void* const* d_in, const int* in_sizes, int n_in,
                              void* d_out, int out_size, void* d_ws, size_t ws_size,
                              hipStream_t stream) {
    const float*     span_hidden = (const float*)d_in[0];
    const float*     span_output = (const float*)d_in[1];
    const float*     nb_output   = (const float*)d_in[2];
    const int*       span_mask   = (const int*)d_in[3];
    const int*       nb_mask     = (const int*)d_in[4];
    const long long* graph_map   = (const long long*)d_in[5];
    const float*     proj_W      = (const float*)d_in[6];
    const float*     proj_b      = (const float*)d_in[7];
    const float*     ws_W        = (const float*)d_in[8];
    const float*     ws_b        = (const float*)d_in[9];
    const float*     gat_W       = (const float*)d_in[10];
    const float*     gat_b       = (const float*)d_in[11];
    const float*     ff_W        = (const float*)d_in[12];
    const float*     ff_b        = (const float*)d_in[13];

    char* w = (char*)d_ws;
    auto carve = [&](size_t bytes) -> char* {
        char* p = w; w += (bytes + 255) & ~(size_t)255; return p;
    };
    half_t* projWt = (half_t*)carve(512 * 512 * sizeof(half_t));
    half_t* wsWt   = (half_t*)carve(2 * 512 * 512 * sizeof(half_t));
    half_t* gatWt  = (half_t*)carve(2 * 8 * 64 * 128 * sizeof(half_t));
    half_t* ffWt   = (half_t*)carve(512 * 1024 * sizeof(half_t));
    half_t* xbar   = (half_t*)carve((size_t)Gsz * 512 * sizeof(half_t));
    half_t* allh   = (half_t*)carve((size_t)Gsz * 512 * sizeof(half_t));
    half_t* chh    = (half_t*)carve((size_t)Gsz * 512 * sizeof(half_t));
    half_t* combh  = (half_t*)carve((size_t)Gsz * 1024 * sizeof(half_t));
    half_t* c1h    = (half_t*)carve((size_t)Gsz * 512 * sizeof(half_t));
    half_t* c2h    = (half_t*)carve((size_t)Bsz * 512 * sizeof(half_t));
    half_t* ffin   = (half_t*)carve((size_t)Bsz * 1024 * sizeof(half_t));
    int*    emptyF = (int*)carve((size_t)Gsz * sizeof(int));

    // 1) weights -> f16 (pre-transposed)
    convert_weights<<<5632, 256, 0, stream>>>(proj_W, ws_W, gat_W, ff_W,
                                              projWt, wsWt, gatWt, ffWt);
    // 2) masked mean (linearity: fold projection after reduction)
    masked_mean_kernel<<<Gsz, 256, 0, stream>>>(span_output, span_mask,
                                                nb_output, nb_mask, xbar);
    // 3) all_h = lrelu(xbar @ proj_W + proj_b)   [G,512]x[512,512]
    gemm_wmma<<<dim3(512 / 64, Gsz / 128, 1), 128, 0, stream>>>(
        xbar, 512, 0, projWt, 0, proj_b, 0,
        allh, nullptr, 512, 0, nullptr, Gsz, 512, 512, ACT_LRELU);

    // ---- hop 0 ----
    gemm_wmma<<<dim3(512 / 64, Gsz / 128, 1), 128, 0, stream>>>(
        allh, 512, 0, wsWt, 0, ws_b, 0,
        chh, nullptr, 512, 0, nullptr, Gsz, 512, 512, ACT_LRELU);
    attn_kernel<<<Gsz, 256, 0, stream>>>(chh, graph_map, combh, emptyF, Gsz);
    gemm_wmma<<<dim3(1, Gsz / 128, 8), 128, 0, stream>>>(
        combh, 1024, 128, gatWt, 64 * 128, gat_b, 64,
        c1h, nullptr, 512, 64, emptyF, Gsz, 64, 128, ACT_TANH);

    // ---- hop 1 ----
    gemm_wmma<<<dim3(512 / 64, Gsz / 128, 1), 128, 0, stream>>>(
        c1h, 512, 0, wsWt + 512 * 512, 0, ws_b + 512, 0,
        chh, nullptr, 512, 0, nullptr, Gsz, 512, 512, ACT_LRELU);
    attn_kernel<<<Bsz, 256, 0, stream>>>(chh, graph_map, combh, emptyF, Bsz);
    gemm_wmma<<<dim3(1, Bsz / 128, 8), 128, 0, stream>>>(
        combh, 1024, 128, gatWt + 8 * 64 * 128, 64 * 128, gat_b + 512, 64,
        c2h, nullptr, 512, 64, emptyF, Bsz, 64, 128, ACT_TANH);

    // 4) ff input: [span_hidden | where(empty, b_hidden, c)]
    concat_kernel<<<(Bsz * 1024) / 256, 256, 0, stream>>>(
        span_hidden, allh, c2h, emptyF, ffin);
    // 5) out = lrelu(ffin @ ff_W + ff_b)  [B,1024]x[1024,512] -> f32 d_out
    gemm_wmma<<<dim3(512 / 64, Bsz / 128, 1), 128, 0, stream>>>(
        ffin, 1024, 0, ffWt, 0, ff_b, 0,
        nullptr, (float*)d_out, 512, 0, nullptr, Bsz, 512, 1024, ACT_LRELU);
}
